// GASADecoderLayer_26173530701969
// MI455X (gfx1250) — matrix operand order repaired and checked by reading, not compiled, not run on hardware
//
#include <hip/hip_runtime.h>

// Flip to 0 if the gfx1250 assembler rejects the async-to-LDS mnemonics.
#ifndef USE_ASYNC_LDS
#define USE_ASYNC_LDS 1
#endif

// ---------------- CDNA5 WMMA types ----------------
typedef _Float16 half8  __attribute__((ext_vector_type(8)));
typedef _Float16 half16 __attribute__((ext_vector_type(16)));
typedef float    floatx8 __attribute__((ext_vector_type(8)));

// ---------------- problem constants ----------------
constexpr int B_  = 4;
constexpr int Qn  = 600;
constexpr int Ln  = 2048;
constexpr int Dm  = 256;
constexpr int Hh  = 8;
constexpr int HDm = 32;
constexpr int DFF = 2048;
constexpr int KDm = 32;
constexpr int NQP = 608;            // Q padded to multiple of 32 (attention)
constexpr int MQ  = B_ * Qn;        // 2400
constexpr int MQP = 2560;           // MQ padded to multiple of 256 (GEMM block tile)
constexpr int ML  = B_ * Ln;        // 8192 (already multiple of 256)
constexpr float SCALE = 0.17677669529663687f; // 32^-0.5

// ---------------- elementwise helpers ----------------
__global__ void k_cvt(const float* __restrict__ s, _Float16* __restrict__ d, int n) {
  int i = blockIdx.x * 256 + threadIdx.x;
  if (i < n) d[i] = (_Float16)s[i];
}

__global__ void k_fill0(_Float16* __restrict__ d, int n) {
  int i = blockIdx.x * 256 + threadIdx.x;
  if (i < n) d[i] = (_Float16)0.0f;
}

// ---------------- WMMA GEMM: C[M,N] = A[M,K] * W[N,K]^T + bias ----------------
// Block = 8 waves, block tile 256x64; wave tile 32x64 (2 M-subtiles x 4 N-subtiles
// = 8 WMMA per K-step). The shared 64x32 W tile is staged into double-buffered
// LDS (fragment-ordered, one aligned 32B read per lane) via CDNA5
// global_load_async_to_lds_b128 (ASYNCcnt), overlapped with compute. All four
// B fragments are pulled from LDS into registers before the 8-WMMA burst so
// the ds_loads clause together instead of draining dscnt per WMMA pair.
// modes: 0 f32 [M,N]; 1 f16 [M,N]; 2 f16 heads [b,h,n,hd]; 3 f16 heads-T [b,h,hd,n].
__global__ __launch_bounds__(256) void k_gemm(
    const _Float16* __restrict__ A, const _Float16* __restrict__ W,
    const float* __restrict__ bias, float* __restrict__ outF,
    _Float16* __restrict__ outH, int M, int N, int K,
    int mode, int relu, int tokens, int pitch) {
  __shared__ _Float16 sB[2][64 * 32];   // 2 x 4KB, fragment-ordered
  const int tid  = threadIdx.x;
  const int lane = tid & 31;
  const int wid  = tid >> 5;
  const int l    = lane & 15;
  const bool hiL = lane >= 16;
  const int mbase = blockIdx.x * 256 + wid * 32;
  const int nb    = blockIdx.y * 64;

  // staging map: 256 threads <-> 256 16-byte chunks of the 64x32 W tile,
  // stored so fragment j for lane L is contiguous at (j*32+L)*16 halves.
  const int slot   = tid >> 1;                  // 0..127 fragment slot
  const int halfc  = tid & 1;                   // which 16B half of the 32B frag
  const int js     = slot >> 5;                 // N-subtile 0..3
  const int lane_s = slot & 31;
  const int ls     = lane_s & 15;
  const int koffs  = ((lane_s >= 16) ? 8 : 0) + halfc * 16;
  const _Float16* wsrc = W + (size_t)(nb + js * 16 + ls) * K + koffs;
  const int ldst = (js * 32 + lane_s) * 16 + halfc * 8;

  const _Float16* arow0 = A + (size_t)(mbase + l) * K + (hiL ? 8 : 0);
  const _Float16* arow1 = A + (size_t)(mbase + 16 + l) * K + (hiL ? 8 : 0);

  floatx8 acc[2][4] = {{floatx8{}, floatx8{}, floatx8{}, floatx8{}},
                       {floatx8{}, floatx8{}, floatx8{}, floatx8{}}};

  auto stage = [&](int kb, int buf) {
#if USE_ASYNC_LDS
    const unsigned lds_off = (unsigned)(size_t)(&sB[buf][ldst]);
    const unsigned long long ga = (unsigned long long)(size_t)(wsrc + kb);
    asm volatile("global_load_async_to_lds_b128 %0, %1, off"
                 :: "v"(lds_off), "v"(ga) : "memory");
#else
    *(half8*)&sB[buf][ldst] = *(const half8*)(wsrc + kb);
#endif
  };
  auto wait_stage = [&]() {
#if USE_ASYNC_LDS
    asm volatile("s_wait_asynccnt 0" ::: "memory");
#endif
    __syncthreads();
  };
  auto loadA = [&](int kb, half16* a) {
    const half8 x0 = *(const half8*)(arow0 + kb);
    const half8 x1 = *(const half8*)(arow0 + kb + 16);
    const half8 y0 = *(const half8*)(arow1 + kb);
    const half8 y1 = *(const half8*)(arow1 + kb + 16);
#pragma unroll
    for (int i = 0; i < 8; ++i) {
      a[0][i] = x0[i]; a[0][8 + i] = x1[i];
      a[1][i] = y0[i]; a[1][8 + i] = y1[i];
    }
  };

  stage(0, 0);
  half16 acur[2];
  loadA(0, acur);
  wait_stage();

  for (int kb = 0; kb < K; kb += 32) {
    const int buf = (kb >> 5) & 1;
    const bool more = (kb + 32) < K;
    if (more) stage(kb + 32, buf ^ 1);            // async fill of other buffer
    half16 anext[2];
    if (more) loadA(kb + 32, anext);              // pipelined A fetch
    if (kb + 128 < K) {
      __builtin_prefetch(arow0 + kb + 128, 0, 1);
      __builtin_prefetch(arow1 + kb + 128, 0, 1);
    }
    // pull all four B fragments first so ds_loads clause together
    half16 bf[4];
#pragma unroll
    for (int j = 0; j < 4; ++j)
      bf[j] = *(const half16*)&sB[buf][(j * 32 + lane) * 16];
#pragma unroll
    for (int j = 0; j < 4; ++j) {
      acc[0][j] = __builtin_amdgcn_wmma_f32_16x16x32_f16(false, acur[0], false, bf[j],
                                                         (short)0, acc[0][j], false, false);
      acc[1][j] = __builtin_amdgcn_wmma_f32_16x16x32_f16(false, acur[1], false, bf[j],
                                                         (short)0, acc[1][j], false, false);
    }
    if (more) { acur[0] = anext[0]; acur[1] = anext[1]; }
    wait_stage();
  }

  // epilogue: C layout lane<16 rows +0..7 / lane>=16 rows +8..15, per M-subtile
#pragma unroll
  for (int s = 0; s < 2; ++s) {
#pragma unroll
    for (int j = 0; j < 4; ++j) {
      const int col = nb + j * 16 + l;
      const float bb = bias ? bias[col] : 0.0f;
#pragma unroll
      for (int r = 0; r < 8; ++r) {
        const int row = mbase + s * 16 + (hiL ? 8 : 0) + r;
        if (row >= M) continue;
        float v = acc[s][j][r] + bb;
        if (relu) v = fmaxf(v, 0.0f);
        if (mode == 0) {
          outF[(size_t)row * N + col] = v;
        } else if (mode == 1) {
          outH[(size_t)row * N + col] = (_Float16)v;
        } else {
          const int bidx = row / tokens;
          const int n    = row - bidx * tokens;
          const int h    = col >> 5;
          const int hd   = col & 31;
          if (mode == 2)
            outH[((size_t)(bidx * Hh + h) * pitch + n) * HDm + hd] = (_Float16)v;
          else
            outH[((size_t)(bidx * Hh + h) * HDm + hd) * pitch + n] = (_Float16)v;
        }
      }
    }
  }
}

// ---------------- GASA geometric bias: clamp(MLP(dist),-10,0), per (b,q,k) ----
__global__ void k_bias(const float* __restrict__ qpos, const float* __restrict__ mpos,
                       const float* __restrict__ w1, const float* __restrict__ b1,
                       const float* __restrict__ w2, const float* __restrict__ b2,
                       float* __restrict__ out) {
  const size_t i = (size_t)blockIdx.x * 256 + threadIdx.x;
  const size_t total = (size_t)B_ * Qn * Ln;
  if (i >= total) return;
  const int k = (int)(i % Ln);
  const size_t t = i / Ln;
  const int q = (int)(t % Qn);
  const int b = (int)(t / Qn);
  const float* qp = qpos + ((size_t)b * Qn + q) * 3;
  const float* mp = mpos + ((size_t)b * Ln + k) * 3;
  const float dx = qp[0] - mp[0], dy = qp[1] - mp[1], dz = qp[2] - mp[2];
  const float dist = sqrtf(fmaxf(dx * dx + dy * dy + dz * dz, 0.0f));
  float acc = b2[0];
#pragma unroll
  for (int j = 0; j < KDm; ++j)
    acc += fmaxf(dist * w1[j] + b1[j], 0.0f) * w2[j];
  acc = fminf(fmaxf(acc, -10.0f), 0.0f);
  out[((size_t)(b * NQP + q)) * Ln + k] = acc;
}

// ---------------- flash attention, one wave per (b,h,16-query tile) ----------
// S^T = K*Q^T so queries sit in lane columns (shfl_xor softmax) and the P^T
// C-layout is directly the A operand of P*V. K fragments for the next chunk
// and V fragments for this chunk are issued before/around the score WMMAs so
// they overlap the exp/softmax VALU work.
__global__ __launch_bounds__(256) void k_attn(
    const _Float16* __restrict__ Qh, const _Float16* __restrict__ Kh,
    const _Float16* __restrict__ Vt, const float* __restrict__ bias,
    const float* __restrict__ betaPtr, _Float16* __restrict__ outT,
    int NQ, int NK, int nqp, int nkp, float scale) {
  const int lane = threadIdx.x & 31;
  const int wid  = threadIdx.x >> 5;
  const int QT   = nqp / 16;
  int w = blockIdx.x * 8 + wid;
  const int total = B_ * Hh * QT;
  if (w >= total) return;                         // wave-uniform, no barriers here
  const int qt = w % QT; w /= QT;
  const int h  = w % Hh;
  const int b  = w / Hh;
  const int bh = b * Hh + h;
  const int l  = lane & 15;
  const bool hiL = lane >= 16;
  const int qbase = qt * 16;
  const float bet = betaPtr ? betaPtr[0] : 0.0f;

  const half16 qb = *(const half16*)(Qh + ((size_t)bh * nqp + qbase + l) * HDm + (hiL ? 16 : 0));
  const float* brow = bias ? (bias + ((size_t)(b * nqp) + qbase + l) * (size_t)nkp) : nullptr;
  const _Float16* vrow0 = Vt + ((size_t)bh * HDm + l) * nkp + (hiL ? 8 : 0);
  const _Float16* vrow1 = Vt + ((size_t)bh * HDm + 16 + l) * nkp + (hiL ? 8 : 0);

  auto loadK = [&](int kb, half16* a) {
    const _Float16* kr0 = Kh + ((size_t)bh * nkp + kb + l) * HDm + (hiL ? 16 : 0);
    const _Float16* kr1 = Kh + ((size_t)bh * nkp + kb + 16 + l) * HDm + (hiL ? 16 : 0);
    const half8 t0 = *(const half8*)(kr0);
    const half8 t1 = *(const half8*)(kr0 + 8);
    const half8 t2 = *(const half8*)(kr1);
    const half8 t3 = *(const half8*)(kr1 + 8);
#pragma unroll
    for (int i = 0; i < 8; ++i) {
      a[0][i] = t0[i]; a[0][8 + i] = t1[i];
      a[1][i] = t2[i]; a[1][8 + i] = t3[i];
    }
  };

  floatx8 o0{}, o1{};
  float mrun = -3.0e38f, lrun = 0.0f;

  half16 acur[2];
  loadK(0, acur);

  for (int kb = 0; kb < nkp; kb += 32) {
    const bool more = (kb + 32) < nkp;
    // V fragments for this chunk: issue early, consumed after softmax
    const half8 v0a = *(const half8*)(vrow0 + kb), v0b = *(const half8*)(vrow0 + kb + 16);
    const half8 v1a = *(const half8*)(vrow1 + kb), v1b = *(const half8*)(vrow1 + kb + 16);
    const half16 v0 = __builtin_shufflevector(v0a, v0b, 0, 1, 2, 3, 4, 5, 6, 7, 8, 9, 10, 11, 12, 13, 14, 15);
    const half16 v1 = __builtin_shufflevector(v1a, v1b, 0, 1, 2, 3, 4, 5, 6, 7, 8, 9, 10, 11, 12, 13, 14, 15);
    if (brow && more) __builtin_prefetch(brow + kb + 32, 0, 1);

    floatx8 z{};
    floatx8 s0 = __builtin_amdgcn_wmma_f32_16x16x32_f16(false, acur[0], false, qb, (short)0, z, false, false);
    floatx8 s1 = __builtin_amdgcn_wmma_f32_16x16x32_f16(false, acur[1], false, qb, (short)0, z, false, false);

    half16 anext[2];
    if (more) loadK(kb + 32, anext);              // overlap with softmax VALU

    float sv0[8], sv1[8];
#pragma unroll
    for (int r = 0; r < 8; ++r) {
      const int k0 = kb + r + (hiL ? 8 : 0);
      const int k1 = k0 + 16;
      float x0 = s0[r] * scale;
      float x1 = s1[r] * scale;
      if (brow) { x0 += bet * brow[k0]; x1 += bet * brow[k1]; }
      if (k0 >= NK) x0 = -1.0e30f;
      if (k1 >= NK) x1 = -1.0e30f;
      sv0[r] = x0; sv1[r] = x1;
    }
    float mloc = sv0[0];
#pragma unroll
    for (int r = 0; r < 8; ++r) { mloc = fmaxf(mloc, sv0[r]); mloc = fmaxf(mloc, sv1[r]); }
    mloc = fmaxf(mloc, __shfl_xor(mloc, 16));
    const float mnew = fmaxf(mrun, mloc);
    const float corr = __expf(mrun - mnew);
#pragma unroll
    for (int r = 0; r < 8; ++r) {
      const float cr = __shfl(corr, (hiL ? 8 : 0) + r);
      o0[r] *= cr; o1[r] *= cr;
    }
    half16 pa;
    float ls = 0.0f;
#pragma unroll
    for (int r = 0; r < 8; ++r) {
      const float p0 = __expf(sv0[r] - mnew);
      const float p1 = __expf(sv1[r] - mnew);
      ls += p0 + p1;
      pa[r] = (_Float16)p0;
      pa[8 + r] = (_Float16)p1;
    }
    ls += __shfl_xor(ls, 16);
    lrun = lrun * corr + ls;
    mrun = mnew;
    o0 = __builtin_amdgcn_wmma_f32_16x16x32_f16(false, pa, false, v0, (short)0, o0, false, false);
    o1 = __builtin_amdgcn_wmma_f32_16x16x32_f16(false, pa, false, v1, (short)0, o1, false, false);
    if (more) { acur[0] = anext[0]; acur[1] = anext[1]; }
  }
#pragma unroll
  for (int r = 0; r < 8; ++r) {
    const int qrow = (hiL ? 8 : 0) + r;
    const float denom = __shfl(lrun, qrow);
    const int qg = qbase + qrow;
    if (qg < NQ) {
      const float inv = 1.0f / denom;
      const size_t o = ((size_t)(b * NQ + qg)) * Dm + h * HDm;
      outT[o + l]      = (_Float16)(o0[r] * inv);
      outT[o + 16 + l] = (_Float16)(o1[r] * inv);
    }
  }
}

// ---------------- residual add + LayerNorm (D=256, one row/block) -----------
__global__ void k_ln(const float* __restrict__ res, const float* __restrict__ del,
                     const float* __restrict__ g, const float* __restrict__ bb,
                     float* __restrict__ out32, _Float16* __restrict__ out16, int rows) {
  __shared__ float sm[256];
  const int row = blockIdx.x;
  const int t = threadIdx.x;
  if (row >= rows) return;
  const size_t o = (size_t)row * Dm + t;
  const float v = res[o] + del[o];
  sm[t] = v;
  __syncthreads();
  for (int s = 128; s > 0; s >>= 1) { if (t < s) sm[t] += sm[t + s]; __syncthreads(); }
  const float mean = sm[0] * (1.0f / Dm);
  __syncthreads();
  const float d = v - mean;
  sm[t] = d * d;
  __syncthreads();
  for (int s = 128; s > 0; s >>= 1) { if (t < s) sm[t] += sm[t + s]; __syncthreads(); }
  const float var = sm[0] * (1.0f / Dm);
  const float y = d * rsqrtf(var + 1e-5f) * g[t] + bb[t];
  out32[o] = y;
  if (out16) out16[o] = (_Float16)y;
}

// ---------------- orchestration ----------------
extern "C" void kernel_launch(void* const* d_in, const int* in_sizes, int n_in,
                              void* d_out, int out_size, void* d_ws, size_t ws_size,
                              hipStream_t stream) {
  (void)in_sizes; (void)n_in; (void)out_size; (void)ws_size;
  const float* queries    = (const float*)d_in[0];
  const float* memory     = (const float*)d_in[1];
  const float* memory_pos = (const float*)d_in[2];
  const float* query_pos  = (const float*)d_in[3];
  const float* sa_in_w  = (const float*)d_in[4];
  const float* sa_in_b  = (const float*)d_in[5];
  const float* sa_out_w = (const float*)d_in[6];
  const float* sa_out_b = (const float*)d_in[7];
  const float* norm1_g  = (const float*)d_in[8];
  const float* norm1_b  = (const float*)d_in[9];
  const float* q_w = (const float*)d_in[10]; const float* q_b = (const float*)d_in[11];
  const float* k_w = (const float*)d_in[12]; const float* k_b = (const float*)d_in[13];
  const float* v_w = (const float*)d_in[14]; const float* v_b = (const float*)d_in[15];
  const float* o_w = (const float*)d_in[16]; const float* o_b = (const float*)d_in[17];
  const float* norm2_g = (const float*)d_in[18];
  const float* norm2_b = (const float*)d_in[19];
  const float* beta    = (const float*)d_in[20];
  const float* dk_w1 = (const float*)d_in[21]; const float* dk_b1 = (const float*)d_in[22];
  const float* dk_w2 = (const float*)d_in[23]; const float* dk_b2 = (const float*)d_in[24];
  const float* ffn_w1 = (const float*)d_in[25]; const float* ffn_b1 = (const float*)d_in[26];
  const float* ffn_w2 = (const float*)d_in[27]; const float* ffn_b2 = (const float*)d_in[28];
  const float* norm3_g = (const float*)d_in[29];
  const float* norm3_b = (const float*)d_in[30];

  // workspace carve-up (256B aligned)
  char* base = (char*)d_ws;
  size_t off = 0;
  auto alloc = [&](size_t bytes) -> void* {
    void* p = base + off;
    off += (bytes + 255) & ~(size_t)255;
    return p;
  };
  const int HB_Q = B_ * Hh * NQP * HDm;   // padded query-side head buffer elems
  const int HB_L = B_ * Hh * Ln * HDm;    // memory-side head buffer elems

  _Float16* w16_sain  = (_Float16*)alloc((size_t)3 * Dm * Dm * 2);
  _Float16* w16_saout = (_Float16*)alloc((size_t)Dm * Dm * 2);
  _Float16* w16_q = (_Float16*)alloc((size_t)Dm * Dm * 2);
  _Float16* w16_k = (_Float16*)alloc((size_t)Dm * Dm * 2);
  _Float16* w16_v = (_Float16*)alloc((size_t)Dm * Dm * 2);
  _Float16* w16_o = (_Float16*)alloc((size_t)Dm * Dm * 2);
  _Float16* w16_f1 = (_Float16*)alloc((size_t)DFF * Dm * 2);
  _Float16* w16_f2 = (_Float16*)alloc((size_t)Dm * DFF * 2);
  // activation buffers used as GEMM "A" are padded to MQP rows
  _Float16* q16 = (_Float16*)alloc((size_t)MQP * Dm * 2);
  _Float16* m16 = (_Float16*)alloc((size_t)ML * Dm * 2);
  _Float16* qh_sa = (_Float16*)alloc((size_t)HB_Q * 2);
  _Float16* kh_sa = (_Float16*)alloc((size_t)HB_Q * 2);
  _Float16* vt_sa = (_Float16*)alloc((size_t)HB_Q * 2);
  _Float16* sacat16 = (_Float16*)alloc((size_t)MQP * Dm * 2);
  _Float16* x1_16 = (_Float16*)alloc((size_t)MQP * Dm * 2);
  _Float16* qh_c = (_Float16*)alloc((size_t)HB_Q * 2);
  _Float16* kh_c = (_Float16*)alloc((size_t)HB_L * 2);
  _Float16* vt_c = (_Float16*)alloc((size_t)HB_L * 2);
  _Float16* cacat16 = (_Float16*)alloc((size_t)MQP * Dm * 2);
  _Float16* x2_16 = (_Float16*)alloc((size_t)MQP * Dm * 2);
  _Float16* h16buf = (_Float16*)alloc((size_t)MQP * DFF * 2);
  float* saout  = (float*)alloc((size_t)MQ * Dm * 4);
  float* x1     = (float*)alloc((size_t)MQ * Dm * 4);
  float* caout  = (float*)alloc((size_t)MQ * Dm * 4);
  float* x2     = (float*)alloc((size_t)MQ * Dm * 4);
  float* ffnout = (float*)alloc((size_t)MQ * Dm * 4);
  float* biasbuf = (float*)alloc((size_t)B_ * NQP * Ln * 4);

  auto cvt = [&](const float* s, _Float16* d, int n) {
    k_cvt<<<(n + 255) / 256, 256, 0, stream>>>(s, d, n);
  };
  auto gemm = [&](const _Float16* A, const _Float16* W, const float* bias,
                  float* oF, _Float16* oH, int M, int N, int K,
                  int mode, int relu, int tokens, int pitch) {
    dim3 grid((M + 255) / 256, N / 64);
    k_gemm<<<grid, 256, 0, stream>>>(A, W, bias, oF, oH, M, N, K, mode, relu, tokens, pitch);
  };
  auto attn = [&](const _Float16* Qh, const _Float16* Kh, const _Float16* Vt,
                  const float* bias, const float* bet, _Float16* out,
                  int NQ, int NK, int nqp, int nkp) {
    const int waves = B_ * Hh * (nqp / 16);
    k_attn<<<(waves + 7) / 8, 256, 0, stream>>>(Qh, Kh, Vt, bias, bet, out,
                                                NQ, NK, nqp, nkp, SCALE);
  };
  auto ln = [&](const float* res, const float* del, const float* g, const float* bb,
                float* o32, _Float16* o16) {
    k_ln<<<MQ, 256, 0, stream>>>(res, del, g, bb, o32, o16, MQ);
  };

  // phase 0: one-time conversions + zero-fill padded SA key/value buffers
  cvt(queries, q16, MQ * Dm);
  cvt(memory, m16, ML * Dm);
  cvt(sa_in_w, w16_sain, 3 * Dm * Dm);
  cvt(sa_out_w, w16_saout, Dm * Dm);
  cvt(q_w, w16_q, Dm * Dm);
  cvt(k_w, w16_k, Dm * Dm);
  cvt(v_w, w16_v, Dm * Dm);
  cvt(o_w, w16_o, Dm * Dm);
  cvt(ffn_w1, w16_f1, DFF * Dm);
  cvt(ffn_w2, w16_f2, Dm * DFF);
  k_fill0<<<(HB_Q + 255) / 256, 256, 0, stream>>>(kh_sa, HB_Q);
  k_fill0<<<(HB_Q + 255) / 256, 256, 0, stream>>>(vt_sa, HB_Q);

  // phase 1: self-attention
  gemm(q16, w16_sain,               sa_in_b,          nullptr, qh_sa, MQ, Dm, Dm, 2, 0, Qn, NQP);
  gemm(q16, w16_sain + Dm * Dm,     sa_in_b + Dm,     nullptr, kh_sa, MQ, Dm, Dm, 2, 0, Qn, NQP);
  gemm(q16, w16_sain + 2 * Dm * Dm, sa_in_b + 2 * Dm, nullptr, vt_sa, MQ, Dm, Dm, 3, 0, Qn, NQP);
  attn(qh_sa, kh_sa, vt_sa, nullptr, nullptr, sacat16, Qn, Qn, NQP, NQP);
  gemm(sacat16, w16_saout, sa_out_b, saout, nullptr, MQ, Dm, Dm, 0, 0, 0, 0);
  ln(queries, saout, norm1_g, norm1_b, x1, x1_16);

  // phase 2: GASA cross-attention (bias shared across heads, computed once)
  {
    const size_t tot = (size_t)B_ * Qn * Ln;
    k_bias<<<(unsigned)((tot + 255) / 256), 256, 0, stream>>>(
        query_pos, memory_pos, dk_w1, dk_b1, dk_w2, dk_b2, biasbuf);
  }
  gemm(x1_16, w16_q, q_b, nullptr, qh_c, MQ, Dm, Dm, 2, 0, Qn, NQP);
  gemm(m16,   w16_k, k_b, nullptr, kh_c, ML, Dm, Dm, 2, 0, Ln, Ln);
  gemm(m16,   w16_v, v_b, nullptr, vt_c, ML, Dm, Dm, 3, 0, Ln, Ln);
  attn(qh_c, kh_c, vt_c, biasbuf, beta, cacat16, Qn, Ln, NQP, Ln);
  gemm(cacat16, w16_o, o_b, caout, nullptr, MQ, Dm, Dm, 0, 0, 0, 0);
  ln(x1, caout, norm2_g, norm2_b, x2, x2_16);

  // phase 3: FFN
  gemm(x2_16, w16_f1, ffn_b1, nullptr, h16buf, MQ, DFF, Dm, 1, 1, 0, 0);
  gemm(h16buf, w16_f2, ffn_b2, ffnout, nullptr, MQ, Dm, DFF, 0, 0, 0, 0);
  ln(x2, ffnout, norm3_g, norm3_b, (float*)d_out, nullptr);
}
